// TimeAwareMultiHeadLatentAttention_50680614093532
// MI455X (gfx1250) — compile-verified, matrix operand order checked by reading, not run on hardware
//
#include <hip/hip_runtime.h>

// Problem constants (reference: TimeAwareMultiHeadLatentAttention)
#define B_    2
#define S_    2048
#define D_    1024
#define H_    16
#define HD_   64
#define L_    256
#define NTOK_ (B_ * S_)

typedef __attribute__((ext_vector_type(16))) _Float16 v16h;
typedef __attribute__((ext_vector_type(8)))  _Float16 v8h;
typedef __attribute__((ext_vector_type(8)))  float    v8f;

static __device__ __forceinline__ v8f wmma_f16f32(v16h a, v16h b, v8f c) {
  // D = A(16x32 f16) * B(32x16 f16) + C(16x16 f32)
  return __builtin_amdgcn_wmma_f32_16x16x32_f16(false, a, false, b, (short)0, c,
                                                false, false);
}

static __device__ __forceinline__ v8f zero8() {
  v8f z = {0.f, 0.f, 0.f, 0.f, 0.f, 0.f, 0.f, 0.f};
  return z;
}

// Load an A-style fragment (row m = lane&15; K chunks at col0+8g and col0+16+8g)
// from a row-major f16 matrix with leading dimension ld. The same routine loads
// B-fragments from the *transposed* (N-major) operand. Works for global or LDS.
static __device__ __forceinline__ v16h load_frag(const _Float16* __restrict__ base,
                                                 int row0, int ld, int col0,
                                                 int lane) {
  const int g = lane >> 4;
  const int m = lane & 15;
  const _Float16* p = base + (size_t)(row0 + m) * ld + col0 + g * 8;
  union { v16h v; v8h h[2]; } u;
  u.h[0] = *(const v8h*)(p);
  u.h[1] = *(const v8h*)(p + 16);
  return u.v;
}

// LDS byte offset of a __shared__ object (ISA 10.x):
// generic -> addrspace(3) cast yields the 32-bit LDS offset.
static __device__ __forceinline__ unsigned lds_off_of(const void* p) {
  return (unsigned)(unsigned long long)(const __attribute__((address_space(3))) char*)p;
}

// CDNA5 async DMA: memory -> LDS, 16B per lane, tracked by ASYNCcnt.
static __device__ __forceinline__ void async_b128_to_lds(unsigned lds_off,
                                                         const void* gaddr) {
  asm volatile("global_load_async_to_lds_b128 %0, %1, off"
               :: "v"(lds_off), "v"((unsigned long long)gaddr)
               : "memory");
}

// ---------------------------------------------------------------------------
// Elementwise prep kernels
// ---------------------------------------------------------------------------
__global__ void f32_to_f16_kernel(const float* __restrict__ in,
                                  _Float16* __restrict__ out, int nelem) {
  int i = blockIdx.x * 256 + threadIdx.x;
  if (i < nelem) out[i] = (_Float16)in[i];
}

__global__ void add2_to_f16_kernel(const float* __restrict__ a,
                                   const float* __restrict__ b,
                                   _Float16* __restrict__ out, int nelem) {
  int i = blockIdx.x * 256 + threadIdx.x;
  if (i < nelem) out[i] = (_Float16)(a[i] + b[i]);
}

// out[c * R + r] = (f16) in[r * C + c]   (in: R x C f32, out: C x R f16)
__global__ void transpose_f32_to_f16_kernel(const float* __restrict__ in,
                                            _Float16* __restrict__ out,
                                            int R, int C) {
  int i = blockIdx.x * 256 + threadIdx.x;
  if (i < R * C) {
    int r = i / C, c = i % C;
    out[(size_t)c * R + r] = (_Float16)in[i];
  }
}

// ---------------------------------------------------------------------------
// Generic WMMA GEMM: C[M x N] = A[M x K] * B[K x N], B given transposed (N x K).
// One wave computes a 16(M) x 64(N) tile: 4 accumulators, A reused 4x.
// MODE 0: f32 out; 1: f16 out; 2: f32 + f16 out; 3: per-head transposed f16
//         scatter (v_rec -> vT[B][H][64][S]).
// ---------------------------------------------------------------------------
template <int MODE>
__global__ __launch_bounds__(256) void gemm_wmma_kernel(
    const _Float16* __restrict__ A, const _Float16* __restrict__ Bt,
    float* __restrict__ Cf, _Float16* __restrict__ Ch, int M, int N, int K) {
  const int lane = threadIdx.x & 31;
  const int wid  = threadIdx.x >> 5;
  const int tN   = N >> 6;
  const int ntiles = (M >> 4) * tN;
  const int tile = blockIdx.x * 8 + wid;
  if (tile >= ntiles) return;  // wave-uniform: EXEC stays all-ones for WMMA
  const int row0 = (tile / tN) << 4;
  const int col0 = (tile % tN) << 6;
  const int g = lane >> 4;
  const int n = lane & 15;

  v8f acc[4];
#pragma unroll
  for (int f = 0; f < 4; ++f) acc[f] = zero8();

  for (int k0 = 0; k0 < K; k0 += 32) {
    v16h a = load_frag(A, row0, K, k0, lane);
#pragma unroll
    for (int f = 0; f < 4; ++f) {
      v16h b = load_frag(Bt, col0 + f * 16, K, k0, lane);
      acc[f] = wmma_f16f32(a, b, acc[f]);
    }
  }

  if constexpr (MODE == 3) {
    // v_rec element C[tok][col] -> vT[b][h][d][s], 8 consecutive tokens per lane
#pragma unroll
    for (int f = 0; f < 4; ++f) {
      const int col = col0 + f * 16 + n;
      const int hh  = col >> 6;
      const int dd  = col & (HD_ - 1);
      const int tok0 = row0 + g * 8;        // 16-aligned tile: never crosses batch
      const int bb = tok0 >> 11;            // / S_
      const int ss = tok0 & (S_ - 1);
      v8h o8;
#pragma unroll
      for (int r = 0; r < 8; ++r) o8[r] = (_Float16)acc[f][r];
      *(v8h*)(Ch + (size_t)((bb * H_ + hh) * HD_ + dd) * S_ + ss) = o8;
    }
  } else {
#pragma unroll
    for (int f = 0; f < 4; ++f) {
#pragma unroll
      for (int r = 0; r < 8; ++r) {
        const size_t idx = (size_t)(row0 + g * 8 + r) * N + col0 + f * 16 + n;
        if constexpr (MODE == 0 || MODE == 2) Cf[idx] = acc[f][r];
        if constexpr (MODE == 1 || MODE == 2) Ch[idx] = (_Float16)acc[f][r];
      }
    }
  }
}

// ---------------------------------------------------------------------------
// Flash attention with multiplicative time decay.
// One block = 8 waves = 8 consecutive 16-query tiles of ONE (b, h); the block
// cooperatively stages each 32-key K/V tile into LDS exactly once via
// global_load_async_to_lds_b128 (ASYNCcnt), double-buffered so the DMA of
// tile k+1 overlaps the 8 WMMAs + online softmax of tile k.
// ---------------------------------------------------------------------------
__global__ __launch_bounds__(256) void flash_attn_decay_kernel(
    const _Float16* __restrict__ Qh,   // [NTOK_, D_] row-major
    const _Float16* __restrict__ Kh,   // [NTOK_, D_] row-major (k_rec)
    const _Float16* __restrict__ Vt,   // [B_, H_, HD_, S_] (v_rec transposed)
    _Float16* __restrict__ Oh) {       // [NTOK_, D_] row-major
  __shared__ __align__(16) _Float16 Ktile[2][32][64];  // [key][dim]  8 KB
  __shared__ __align__(16) _Float16 Vtile[2][64][32];  // [dim][key]  8 KB
  __shared__ __align__(16) _Float16 pbuf[8][16][32];   // P transpose 8 KB

  const int t    = threadIdx.x;
  const int lane = t & 31;
  const int wid  = t >> 5;
  const int bh   = blockIdx.x >> 4;              // 16 blocks per (b,h)
  const int qt   = ((blockIdx.x & 15) << 3) + wid;
  const int h    = bh & (H_ - 1);
  const int b    = bh >> 4;
  const int g    = lane >> 4;
  const int n    = lane & 15;

  const float scale = 1.0f / (sqrtf((float)D_) + 1e-8f);
  const float lam   = 1e-4f;  // LAMBDA_DECAY * TIME_STEP
  const float CLMP  = 1e4f;

  const int bS    = b * S_;
  const int qrow0 = bS + qt * 16;
  const v16h qa0 = load_frag(Qh, qrow0, D_, h * HD_ + 0,  lane);
  const v16h qa1 = load_frag(Qh, qrow0, D_, h * HD_ + 32, lane);

  const _Float16* vt = Vt + (size_t)(b * H_ + h) * HD_ * S_;

  // Per-thread staging coordinates: 256 threads x 16B = one 4 KB tile each.
  const int krow = t >> 3, kcc = (t & 7) * 8;    // K: [32 keys][64 dims]
  const int vd   = t >> 2, vkc = (t & 3) * 8;    // V: [64 dims][32 keys]

  v8f o[4];
  float mrun[8], lrun[8];
#pragma unroll
  for (int f = 0; f < 4; ++f) o[f] = zero8();
#pragma unroll
  for (int r = 0; r < 8; ++r) { mrun[r] = -3.0e38f; lrun[r] = 0.f; }

  const int NT = S_ / 32;

  // Prologue: stage tile 0 into buffer 0.
  async_b128_to_lds(lds_off_of(&Ktile[0][krow][kcc]),
                    Kh + (size_t)(bS + krow) * D_ + h * HD_ + kcc);
  async_b128_to_lds(lds_off_of(&Vtile[0][vd][vkc]),
                    vt + (size_t)vd * S_ + vkc);

  for (int kb = 0; kb < NT; ++kb) {
    const int cur = kb & 1;
    if (kb + 1 < NT) {
      // Issue next tile's DMA, then wait only for the CURRENT tile (keep the
      // next stage's 2 async ops in flight).
      const int krow1 = bS + (kb + 1) * 32;
      async_b128_to_lds(lds_off_of(&Ktile[cur ^ 1][krow][kcc]),
                        Kh + (size_t)(krow1 + krow) * D_ + h * HD_ + kcc);
      async_b128_to_lds(lds_off_of(&Vtile[cur ^ 1][vd][vkc]),
                        vt + (size_t)vd * S_ + (kb + 1) * 32 + vkc);
      asm volatile("s_wait_asynccnt 0x2" ::: "memory");
    } else {
      asm volatile("s_wait_asynccnt 0x0" ::: "memory");
    }
    __syncthreads();  // current tile visible to all waves

    // ---- scores: 16 queries x 32 keys from LDS tile ----
    const _Float16* kt  = &Ktile[cur][0][0];
    v16h kb00 = load_frag(kt, 0,  HD_, 0,  lane);
    v16h kb01 = load_frag(kt, 0,  HD_, 32, lane);
    v16h kb10 = load_frag(kt, 16, HD_, 0,  lane);
    v16h kb11 = load_frag(kt, 16, HD_, 32, lane);
    v8f s0 = zero8(), s1 = zero8();
    s0 = wmma_f16f32(qa0, kb00, s0);
    s0 = wmma_f16f32(qa1, kb01, s0);
    s1 = wmma_f16f32(qa0, kb10, s1);
    s1 = wmma_f16f32(qa1, kb11, s1);

    // ---- scale, clamp, time decay, clamp ----
    float p0[8], p1[8], rmax[8], rsum[8], alpha[8];
#pragma unroll
    for (int r = 0; r < 8; ++r) {
      const float qi  = (float)(qt * 16 + g * 8 + r);
      const float kj0 = (float)(kb * 32 + n);
      float t0 = fminf(fmaxf(s0[r] * scale, -CLMP), CLMP);
      t0 *= __expf(-lam * fabsf(qi - kj0));
      t0 = fminf(fmaxf(t0, -CLMP), CLMP);
      float t1 = fminf(fmaxf(s1[r] * scale, -CLMP), CLMP);
      t1 *= __expf(-lam * fabsf(qi - (kj0 + 16.0f)));
      t1 = fminf(fmaxf(t1, -CLMP), CLMP);
      p0[r] = t0; p1[r] = t1;
      rmax[r] = fmaxf(t0, t1);
    }

    // ---- online softmax: row reductions across the 16 lanes of each half ----
#pragma unroll
    for (int msk = 1; msk < 16; msk <<= 1)
#pragma unroll
      for (int r = 0; r < 8; ++r)
        rmax[r] = fmaxf(rmax[r], __shfl_xor(rmax[r], msk, 32));
#pragma unroll
    for (int r = 0; r < 8; ++r) {
      const float mnew = fmaxf(mrun[r], rmax[r]);
      alpha[r] = __expf(mrun[r] - mnew);
      mrun[r]  = mnew;
      p0[r] = __expf(p0[r] - mnew);
      p1[r] = __expf(p1[r] - mnew);
      rsum[r] = p0[r] + p1[r];
    }
#pragma unroll
    for (int msk = 1; msk < 16; msk <<= 1)
#pragma unroll
      for (int r = 0; r < 8; ++r)
        rsum[r] += __shfl_xor(rsum[r], msk, 32);
#pragma unroll
    for (int r = 0; r < 8; ++r) lrun[r] = lrun[r] * alpha[r] + rsum[r];
#pragma unroll
    for (int f = 0; f < 4; ++f)
#pragma unroll
      for (int r = 0; r < 8; ++r) o[f][r] *= alpha[r];

    // ---- P: C-layout -> A-layout via LDS (in-order within a wave) ----
#pragma unroll
    for (int r = 0; r < 8; ++r) {
      pbuf[wid][g * 8 + r][n]      = (_Float16)p0[r];
      pbuf[wid][g * 8 + r][16 + n] = (_Float16)p1[r];
    }
    union { v16h v; v8h hh[2]; } up;
    const _Float16* pp = &pbuf[wid][n][0];  // A-frag row m = lane&15
    up.hh[0] = *(const v8h*)(pp + g * 8);
    up.hh[1] = *(const v8h*)(pp + 16 + g * 8);

    // ---- O += P(16x32) @ V(32x64): B-frags contiguous from LDS V tile ----
    const _Float16* vtl = &Vtile[cur][0][0];
#pragma unroll
    for (int f = 0; f < 4; ++f) {
      v16h vb = load_frag(vtl, f * 16, 32, 0, lane);
      o[f] = wmma_f16f32(up.v, vb, o[f]);
    }

    __syncthreads();  // all waves done reading buf before its next overwrite
  }

  // ---- epilogue: normalize and store row-major f16 ----
  float inv[8];
#pragma unroll
  for (int r = 0; r < 8; ++r) inv[r] = 1.0f / lrun[r];
#pragma unroll
  for (int f = 0; f < 4; ++f)
#pragma unroll
    for (int r = 0; r < 8; ++r)
      Oh[(size_t)(qrow0 + g * 8 + r) * D_ + h * HD_ + f * 16 + n] =
          (_Float16)(o[f][r] * inv[r]);
}

// ---------------------------------------------------------------------------
// Host orchestration
// ---------------------------------------------------------------------------
extern "C" void kernel_launch(void* const* d_in, const int* in_sizes, int n_in,
                              void* d_out, int out_size, void* d_ws,
                              size_t ws_size, hipStream_t stream) {
  (void)in_sizes; (void)n_in; (void)out_size; (void)ws_size;
  const float* x     = (const float*)d_in[0];
  const float* Wq    = (const float*)d_in[1];
  const float* Wk    = (const float*)d_in[2];
  const float* Wv    = (const float*)d_in[3];
  const float* Wo    = (const float*)d_in[4];
  const float* Wdown = (const float*)d_in[5];
  const float* Wupk  = (const float*)d_in[6];
  const float* Wupv  = (const float*)d_in[7];
  float* out = (float*)d_out;  // [NTOK_*D_] attn-out, then [NTOK_*L_] latent

  // Carve workspace (256B aligned)
  char* wsp = (char*)d_ws;
  auto carve = [&](size_t bytes) {
    char* p = wsp;
    wsp += (bytes + 255) & ~(size_t)255;
    return p;
  };
  _Float16* x_h      = (_Float16*)carve((size_t)NTOK_ * D_ * 2);
  _Float16* q_h      = (_Float16*)carve((size_t)NTOK_ * D_ * 2);
  _Float16* k_h      = (_Float16*)carve((size_t)NTOK_ * D_ * 2);
  _Float16* vT       = (_Float16*)carve((size_t)NTOK_ * D_ * 2);
  _Float16* attn_h   = (_Float16*)carve((size_t)NTOK_ * D_ * 2);
  _Float16* latent_h = (_Float16*)carve((size_t)NTOK_ * L_ * 2);
  _Float16* wq_t     = (_Float16*)carve((size_t)D_ * D_ * 2);
  _Float16* wo_t     = (_Float16*)carve((size_t)D_ * D_ * 2);
  _Float16* wkv_h    = (_Float16*)carve((size_t)D_ * D_ * 2);
  _Float16* wdown_t  = (_Float16*)carve((size_t)D_ * L_ * 2);
  _Float16* wupk_t   = (_Float16*)carve((size_t)D_ * L_ * 2);
  _Float16* wupv_t   = (_Float16*)carve((size_t)D_ * L_ * 2);
  float*    wcomb_f  = (float*)   carve((size_t)D_ * L_ * 4);
  _Float16* wcomb_t  = (_Float16*)carve((size_t)D_ * L_ * 2);

  auto eblk = [](int nel) { return (nel + 255) / 256; };
  auto gblk = [](int M, int N) { return ((M >> 4) * (N >> 6) + 7) / 8; };

  // --- precision conversion & weight transposes ---
  f32_to_f16_kernel<<<eblk(NTOK_ * D_), 256, 0, stream>>>(x, x_h, NTOK_ * D_);
  add2_to_f16_kernel<<<eblk(D_ * D_), 256, 0, stream>>>(Wk, Wv, wkv_h, D_ * D_);
  transpose_f32_to_f16_kernel<<<eblk(D_ * D_), 256, 0, stream>>>(Wq, wq_t, D_, D_);
  transpose_f32_to_f16_kernel<<<eblk(D_ * D_), 256, 0, stream>>>(Wo, wo_t, D_, D_);
  transpose_f32_to_f16_kernel<<<eblk(D_ * L_), 256, 0, stream>>>(Wdown, wdown_t, D_, L_);
  transpose_f32_to_f16_kernel<<<eblk(L_ * D_), 256, 0, stream>>>(Wupk, wupk_t, L_, D_);
  transpose_f32_to_f16_kernel<<<eblk(L_ * D_), 256, 0, stream>>>(Wupv, wupv_t, L_, D_);

  // --- W_comb = (Wk + Wv) @ W_down  (folds the k/v GEMMs away) ---
  gemm_wmma_kernel<0><<<gblk(D_, L_), 256, 0, stream>>>(
      wkv_h, wdown_t, wcomb_f, nullptr, D_, L_, D_);
  transpose_f32_to_f16_kernel<<<eblk(D_ * L_), 256, 0, stream>>>(
      wcomb_f, wcomb_t, D_, L_);

  // --- latent = x @ W_comb : f32 -> d_out tail, f16 -> latent_h ---
  gemm_wmma_kernel<2><<<gblk(NTOK_, L_), 256, 0, stream>>>(
      x_h, wcomb_t, out + (size_t)NTOK_ * D_, latent_h, NTOK_, L_, D_);

  // --- q = x @ Wq ---
  gemm_wmma_kernel<1><<<gblk(NTOK_, D_), 256, 0, stream>>>(
      x_h, wq_t, nullptr, q_h, NTOK_, D_, D_);

  // --- k_rec = latent @ W_up_k (row-major == B^T for QK^T) ---
  gemm_wmma_kernel<1><<<gblk(NTOK_, D_), 256, 0, stream>>>(
      latent_h, wupk_t, nullptr, k_h, NTOK_, D_, L_);

  // --- v_rec = latent @ W_up_v, written per-head transposed [B,H,64,S] ---
  gemm_wmma_kernel<3><<<gblk(NTOK_, D_), 256, 0, stream>>>(
      latent_h, wupv_t, nullptr, vT, NTOK_, D_, L_);

  // --- flash attention with decay: 512 blocks, 8 q-tiles of one (b,h) each ---
  flash_attn_decay_kernel<<<B_ * H_ * (S_ / 128), 256, 0, stream>>>(
      q_h, k_h, vT, attn_h);

  // --- out = attn @ Wo ---
  gemm_wmma_kernel<0><<<gblk(NTOK_, D_), 256, 0, stream>>>(
      attn_h, wo_t, out, nullptr, NTOK_, D_, D_);
}
